// EventfulEncoderLayer_37383395344637
// MI455X (gfx1250) — matrix operand order repaired and checked by reading, not compiled, not run on hardware
//
#include <hip/hip_runtime.h>

#define B_  32
#define N_  257
#define D_  1024
#define H_  16
#define HD_ 64
#define FF_ 4096
#define M_  (B_ * N_)   // 8224 rows

typedef __bf16 bf16x16 __attribute__((ext_vector_type(16)));
typedef float  f32x8   __attribute__((ext_vector_type(8)));

union FragAB { bf16x16 v; uint4 q[2]; unsigned short s[16]; };
union FragC  { f32x8  v; float f[8]; };

__device__ __forceinline__ unsigned short f2bf(float f) {
    unsigned int u = __float_as_uint(f);
    u += 0x7fffu + ((u >> 16) & 1u);   // round-to-nearest-even
    return (unsigned short)(u >> 16);
}

// ---- CDNA5 async global->LDS copy (ASYNCcnt-tracked, no VGPR round trip) ----
__device__ __forceinline__ void async_ld16(void* lds_dst, const void* gsrc) {
    asm volatile("global_load_async_to_lds_b128 %0, %1, off"
                 :: "v"((unsigned)(size_t)lds_dst), "v"(gsrc)
                 : "memory");
}
__device__ __forceinline__ void wait_async0() {
    asm volatile("s_wait_asynccnt 0x0" ::: "memory");
}

// ---------------------------------------------------------------- utilities
__global__ void cvt_f32_bf16(const float* __restrict__ s, unsigned short* __restrict__ d, int n) {
    for (int i = blockIdx.x * blockDim.x + threadIdx.x; i < n; i += gridDim.x * blockDim.x)
        d[i] = f2bf(s[i]);
}

// LayerNorm over D=1024, one block (256 thr) per row, bf16 output
__global__ __launch_bounds__(256) void layernorm_to_bf16(
    const float* __restrict__ x, const float* __restrict__ g, const float* __restrict__ bb,
    unsigned short* __restrict__ out)
{
    __shared__ float red[256];
    const int row = blockIdx.x, tid = threadIdx.x;
    const float* p = x + (size_t)row * D_;
    float v[4]; float s = 0.f;
#pragma unroll
    for (int j = 0; j < 4; j++) { v[j] = p[tid + j * 256]; s += v[j]; }
    red[tid] = s; __syncthreads();
    for (int o = 128; o > 0; o >>= 1) { if (tid < o) red[tid] += red[tid + o]; __syncthreads(); }
    const float mu = red[0] * (1.f / D_);
    __syncthreads();
    float ss = 0.f;
#pragma unroll
    for (int j = 0; j < 4; j++) { float d = v[j] - mu; ss += d * d; }
    red[tid] = ss; __syncthreads();
    for (int o = 128; o > 0; o >>= 1) { if (tid < o) red[tid] += red[tid + o]; __syncthreads(); }
    const float rstd = rsqrtf(red[0] * (1.f / D_) + 1e-5f);
    unsigned short* q = out + (size_t)row * D_;
#pragma unroll
    for (int j = 0; j < 4; j++) {
        int col = tid + j * 256;
        q[col] = f2bf((v[j] - mu) * rstd * g[col] + bb[col]);
    }
}

// ------------------------------------------------- generic WMMA bf16 GEMM
// C[M,Nout] = A[M,K] * W[Nout,K]^T + bias, epilogue by MODE:
//   0: outb = bf16((acc+bias)*scale)      (Q/K/V projections)
//   1: outb = bf16(quickgelu(acc+bias))   (FC1)
//   2: outf = acc + bias + resid          (O-proj, FC2)
// Double-buffered LDS, staged with async global->LDS copies overlapped
// against the WMMA stream of the other buffer.
template<int MODE>
__global__ __launch_bounds__(256) void gemm_bf16_wmma(
    const unsigned short* __restrict__ A,
    const unsigned short* __restrict__ W,
    const float* __restrict__ bias,
    const float* __restrict__ resid,
    unsigned short* __restrict__ outb,
    float* __restrict__ outf,
    int M, int K, int Nout, float scale)
{
    __shared__ unsigned short As[2][128 * 72];
    __shared__ unsigned short Bs[2][64 * 72];
    const int tid = threadIdx.x;
    const int lane = tid & 31, w = tid >> 5;
    const int laneLo = lane & 15, laneHi = lane >> 4;
    const int waveRow = (w & 3) * 32, waveCol = (w >> 2) * 32;
    const int mbase = blockIdx.x * 128, nbase = blockIdx.y * 64;

    // per-thread staging chunk geometry (constant across K tiles)
    const int idx0 = tid;                 // A chunks: idx0..idx0+3*256
    auto stage = [&](int buf, int k0) {
#pragma unroll
        for (int i = 0; i < 4; i++) {
            int idx = idx0 + i * 256;             // 1024 x 16B  (A tile 128x64)
            int r = idx >> 3, c = (idx & 7) * 8;
            int gr = mbase + r; if (gr >= M) gr = M - 1;   // clamp: pad rows are store-guarded
            async_ld16(&As[buf][r * 72 + c], A + (size_t)gr * K + k0 + c);
        }
#pragma unroll
        for (int i = 0; i < 2; i++) {
            int idx = idx0 + i * 256;             // 512 x 16B   (W tile 64x64)
            int r = idx >> 3, c = (idx & 7) * 8;
            async_ld16(&Bs[buf][r * 72 + c], W + (size_t)(nbase + r) * K + k0 + c);
        }
    };

    FragC acc[2][2];
#pragma unroll
    for (int i = 0; i < 2; i++)
#pragma unroll
        for (int j = 0; j < 2; j++)
#pragma unroll
            for (int r = 0; r < 8; r++) acc[i][j].f[r] = 0.f;

    stage(0, 0);
    const int nIter = K >> 6;
    for (int it = 0; it < nIter; ++it) {
        const int cur = it & 1;
        wait_async0();          // this wave's copies into As/Bs[cur] complete
        __syncthreads();        // everyone's copies complete
        if (it + 1 < nIter) stage(cur ^ 1, (it + 1) << 6);   // overlap next tile copy

        const unsigned short* AsB = As[cur];
        const unsigned short* BsB = Bs[cur];
#pragma unroll
        for (int ks = 0; ks < 64; ks += 32) {
            FragAB a[2], b[2];
#pragma unroll
            for (int tm = 0; tm < 2; tm++) {
                const unsigned short* p = AsB + (waveRow + tm * 16 + laneLo) * 72 + ks + laneHi * 8;
                a[tm].q[0] = *(const uint4*)(p);        // K {0..7} / {8..15}
                a[tm].q[1] = *(const uint4*)(p + 16);   // K {16..23} / {24..31}
            }
#pragma unroll
            for (int tn = 0; tn < 2; tn++) {
                const unsigned short* p = BsB + (waveCol + tn * 16 + laneLo) * 72 + ks + laneHi * 8;
                b[tn].q[0] = *(const uint4*)(p);        // K {0..7} / {8..15}
                b[tn].q[1] = *(const uint4*)(p + 16);   // K {16..23} / {24..31}
            }
#pragma unroll
            for (int tm = 0; tm < 2; tm++)
#pragma unroll
                for (int tn = 0; tn < 2; tn++)
                    acc[tm][tn].v = __builtin_amdgcn_wmma_f32_16x16x32_bf16(
                        false, a[tm].v, false, b[tn].v, (short)0, acc[tm][tn].v, false, false);
        }
        __syncthreads();        // readers done before buffer is re-staged
    }

#pragma unroll
    for (int tm = 0; tm < 2; tm++)
#pragma unroll
        for (int tn = 0; tn < 2; tn++) {
            const int col = nbase + waveCol + tn * 16 + laneLo;
            const float bv = bias[col];
#pragma unroll
            for (int r = 0; r < 8; r++) {
                const int row = mbase + waveRow + tm * 16 + laneHi * 8 + r;
                if (row < M) {
                    float v = acc[tm][tn].f[r] + bv;
                    size_t o = (size_t)row * Nout + col;
                    if (MODE == 0)      outb[o] = f2bf(v * scale);
                    else if (MODE == 1) outb[o] = f2bf(v / (1.f + __expf(-1.702f * v)));
                    else                outf[o] = v + resid[o];
                }
            }
        }
}

// ------------------------------------------- attention scores  S = q . k^T
// grid (9 qtiles of 32, B*H); q pre-scaled by 1/sqrt(HD); out fp32 [B,H,N,N]
// Row/col overruns are clamped (not branched): K-dim is the full HD=64, so
// clamped rows/cols only produce store-guarded outputs.
__global__ __launch_bounds__(256) void attn_scores(
    const unsigned short* __restrict__ qb, const unsigned short* __restrict__ kbuf,
    float* __restrict__ attn)
{
    const int tid = threadIdx.x, lane = tid & 31, w = tid >> 5;
    const int laneLo = lane & 15, laneHi = lane >> 4;
    const int bh = blockIdx.y, b = bh >> 4, h = bh & 15;
    const int qbase = blockIdx.x * 32;

    for (int kt = w; kt < 9; kt += 8) {            // uniform across the wave
        const int kb0 = kt * 32;
        FragC acc[2][2];
#pragma unroll
        for (int i = 0; i < 2; i++)
#pragma unroll
            for (int j = 0; j < 2; j++)
#pragma unroll
                for (int r = 0; r < 8; r++) acc[i][j].f[r] = 0.f;

#pragma unroll
        for (int ks = 0; ks < HD_; ks += 32) {
            FragAB a[2], bf[2];
#pragma unroll
            for (int tm = 0; tm < 2; tm++) {
                int qr = qbase + tm * 16 + laneLo; if (qr >= N_) qr = N_ - 1;
                const unsigned short* p = qb + ((size_t)(b * N_ + qr) * H_ + h) * HD_ + ks + laneHi * 8;
                a[tm].q[0] = *(const uint4*)(p);
                a[tm].q[1] = *(const uint4*)(p + 16);
            }
#pragma unroll
            for (int tn = 0; tn < 2; tn++) {
                int kr = kb0 + tn * 16 + laneLo; if (kr >= N_) kr = N_ - 1;
                const unsigned short* p = kbuf + ((size_t)(b * N_ + kr) * H_ + h) * HD_ + ks + laneHi * 8;
                bf[tn].q[0] = *(const uint4*)(p);
                bf[tn].q[1] = *(const uint4*)(p + 16);
            }
#pragma unroll
            for (int tm = 0; tm < 2; tm++)
#pragma unroll
                for (int tn = 0; tn < 2; tn++)
                    acc[tm][tn].v = __builtin_amdgcn_wmma_f32_16x16x32_bf16(
                        false, a[tm].v, false, bf[tn].v, (short)0, acc[tm][tn].v, false, false);
        }
#pragma unroll
        for (int tm = 0; tm < 2; tm++)
#pragma unroll
            for (int tn = 0; tn < 2; tn++)
#pragma unroll
                for (int r = 0; r < 8; r++) {
                    const int row = qbase + tm * 16 + laneHi * 8 + r;
                    const int col = kb0 + tn * 16 + laneLo;
                    if (row < N_ && col < N_)
                        attn[((size_t)bh * N_ + row) * N_ + col] = acc[tm][tn].f[r];
                }
    }
}

// row softmax over N=257, in place; one block per (b,h,q) row
__global__ __launch_bounds__(256) void softmax_rows(float* __restrict__ attn)
{
    __shared__ float red[256];
    const int tid = threadIdx.x;
    float* p = attn + (size_t)blockIdx.x * N_;
    float x0 = p[tid];
    float x1 = (tid + 256 < N_) ? p[tid + 256] : -3.4e38f;
    red[tid] = fmaxf(x0, x1); __syncthreads();
    for (int o = 128; o > 0; o >>= 1) { if (tid < o) red[tid] = fmaxf(red[tid], red[tid + o]); __syncthreads(); }
    const float mx = red[0]; __syncthreads();
    float e0 = __expf(x0 - mx);
    float e1 = (tid + 256 < N_) ? __expf(x1 - mx) : 0.f;
    red[tid] = e0 + e1; __syncthreads();
    for (int o = 128; o > 0; o >>= 1) { if (tid < o) red[tid] += red[tid + o]; __syncthreads(); }
    const float inv = 1.f / red[0];
    p[tid] = e0 * inv;
    if (tid + 256 < N_) p[tid + 256] = e1 * inv;
}

// ---------------------------------------------- attn @ V  -> bf16 [M, D]
// grid (3 qtiles of 128, B*H); V staged transposed in LDS per 32-key step.
// Here the key index IS the WMMA K-dim, so out-of-range probs must be zeroed.
__global__ __launch_bounds__(256) void attn_av(
    const float* __restrict__ attn, const unsigned short* __restrict__ vb,
    unsigned short* __restrict__ outb)
{
    __shared__ unsigned short Vt[64 * 40];   // [hd][key], padded stride 40
    const int tid = threadIdx.x, lane = tid & 31, w = tid >> 5;
    const int laneLo = lane & 15, laneHi = lane >> 4;
    const int wm = w & 3, wn = w >> 2;
    const int bh = blockIdx.y, b = bh >> 4, h = bh & 15;
    const int qbase = blockIdx.x * 128;

    FragC acc[2][2];
#pragma unroll
    for (int i = 0; i < 2; i++)
#pragma unroll
        for (int j = 0; j < 2; j++)
#pragma unroll
            for (int r = 0; r < 8; r++) acc[i][j].f[r] = 0.f;

    for (int kb = 0; kb < 288; kb += 32) {
        // stage V[kb..kb+31][0..63] transposed (zeros beyond N: paired probs are zeroed too,
        // but keep LDS clean of stale NaN patterns)
#pragma unroll
        for (int i = 0; i < 8; i++) {
            int idx = tid + i * 256;           // 2048 elements
            int key = idx >> 6, hd = idx & 63;
            int gk = kb + key;
            unsigned short val = (gk < N_) ? vb[((size_t)(b * N_ + gk) * H_ + h) * HD_ + hd] : (unsigned short)0;
            Vt[hd * 40 + key] = val;
        }
        __syncthreads();

        FragAB a[2], bf[2];
#pragma unroll
        for (int tm = 0; tm < 2; tm++) {
            const int qr = qbase + wm * 32 + tm * 16 + laneLo;
            const float* pr = attn + ((size_t)bh * N_ + (qr < N_ ? qr : 0)) * N_;
#pragma unroll
            for (int j = 0; j < 8; j++) {
                int k1 = kb + laneHi * 8 + j;
                int k2 = k1 + 16;
                float v1 = (qr < N_ && k1 < N_) ? pr[k1] : 0.f;
                float v2 = (qr < N_ && k2 < N_) ? pr[k2] : 0.f;
                a[tm].s[j]     = f2bf(v1);
                a[tm].s[8 + j] = f2bf(v2);
            }
        }
#pragma unroll
        for (int tn = 0; tn < 2; tn++) {
            const unsigned short* p = Vt + (wn * 32 + tn * 16 + laneLo) * 40 + laneHi * 8;
            bf[tn].q[0] = *(const uint4*)(p);
            bf[tn].q[1] = *(const uint4*)(p + 16);
        }
#pragma unroll
        for (int tm = 0; tm < 2; tm++)
#pragma unroll
            for (int tn = 0; tn < 2; tn++)
                acc[tm][tn].v = __builtin_amdgcn_wmma_f32_16x16x32_bf16(
                    false, a[tm].v, false, bf[tn].v, (short)0, acc[tm][tn].v, false, false);
        __syncthreads();
    }

#pragma unroll
    for (int tm = 0; tm < 2; tm++)
#pragma unroll
        for (int tn = 0; tn < 2; tn++) {
            const int col = h * HD_ + wn * 32 + tn * 16 + laneLo;
#pragma unroll
            for (int r = 0; r < 8; r++) {
                const int qr = qbase + wm * 32 + tm * 16 + laneHi * 8 + r;
                if (qr < N_)
                    outb[(size_t)(b * N_ + qr) * D_ + col] = f2bf(acc[tm][tn].f[r]);
            }
        }
}

// ---------------------------------------------------------------- launcher
extern "C" void kernel_launch(void* const* d_in, const int* in_sizes, int n_in,
                              void* d_out, int out_size, void* d_ws, size_t ws_size,
                              hipStream_t stream)
{
    const float* hidden = (const float*)d_in[0];
    const float* ln1_g  = (const float*)d_in[1];
    const float* ln1_b  = (const float*)d_in[2];
    const float* q_w    = (const float*)d_in[3];
    const float* q_b    = (const float*)d_in[4];
    const float* k_w    = (const float*)d_in[5];
    const float* k_b    = (const float*)d_in[6];
    const float* v_w    = (const float*)d_in[7];
    const float* v_b    = (const float*)d_in[8];
    const float* o_w    = (const float*)d_in[9];
    const float* o_b    = (const float*)d_in[10];
    const float* ln2_g  = (const float*)d_in[11];
    const float* ln2_b  = (const float*)d_in[12];
    const float* fc1_w  = (const float*)d_in[13];
    const float* fc1_b  = (const float*)d_in[14];
    const float* fc2_w  = (const float*)d_in[15];
    const float* fc2_b  = (const float*)d_in[16];

    char* p = (char*)d_ws;
    auto take = [&](size_t bytes) -> void* {
        void* r = (void*)p; p += (bytes + 255) & ~(size_t)255; return r;
    };
    unsigned short* wqb  = (unsigned short*)take((size_t)D_ * D_ * 2);
    unsigned short* wkb  = (unsigned short*)take((size_t)D_ * D_ * 2);
    unsigned short* wvb  = (unsigned short*)take((size_t)D_ * D_ * 2);
    unsigned short* wob  = (unsigned short*)take((size_t)D_ * D_ * 2);
    unsigned short* wf1b = (unsigned short*)take((size_t)FF_ * D_ * 2);
    unsigned short* wf2b = (unsigned short*)take((size_t)D_ * FF_ * 2);
    unsigned short* xb   = (unsigned short*)take((size_t)M_ * D_ * 2);
    unsigned short* qb   = (unsigned short*)take((size_t)M_ * D_ * 2);
    unsigned short* kb   = (unsigned short*)take((size_t)M_ * D_ * 2);
    unsigned short* vbuf = (unsigned short*)take((size_t)M_ * D_ * 2);
    unsigned short* aob  = (unsigned short*)take((size_t)M_ * D_ * 2);
    unsigned short* yb   = (unsigned short*)take((size_t)M_ * D_ * 2);
    unsigned short* ffb  = (unsigned short*)take((size_t)M_ * FF_ * 2);
    float*          h1   = (float*)take((size_t)M_ * D_ * 4);

    float* outh = (float*)d_out;                 // [B,N,D]
    float* attn = outh + (size_t)M_ * D_;        // [B,H,N,N]

    // weights -> bf16 (L2-resident thereafter)
    cvt_f32_bf16<<<2048, 256, 0, stream>>>(q_w,   wqb,  D_ * D_);
    cvt_f32_bf16<<<2048, 256, 0, stream>>>(k_w,   wkb,  D_ * D_);
    cvt_f32_bf16<<<2048, 256, 0, stream>>>(v_w,   wvb,  D_ * D_);
    cvt_f32_bf16<<<2048, 256, 0, stream>>>(o_w,   wob,  D_ * D_);
    cvt_f32_bf16<<<2048, 256, 0, stream>>>(fc1_w, wf1b, FF_ * D_);
    cvt_f32_bf16<<<2048, 256, 0, stream>>>(fc2_w, wf2b, D_ * FF_);

    layernorm_to_bf16<<<M_, 256, 0, stream>>>(hidden, ln1_g, ln1_b, xb);

    const dim3 gD((M_ + 127) / 128, D_ / 64);    // 65 x 16
    gemm_bf16_wmma<0><<<gD, 256, 0, stream>>>(xb, wqb, q_b, nullptr, qb,   nullptr, M_, D_, D_, 0.125f);
    gemm_bf16_wmma<0><<<gD, 256, 0, stream>>>(xb, wkb, k_b, nullptr, kb,   nullptr, M_, D_, D_, 1.0f);
    gemm_bf16_wmma<0><<<gD, 256, 0, stream>>>(xb, wvb, v_b, nullptr, vbuf, nullptr, M_, D_, D_, 1.0f);

    attn_scores<<<dim3(9, B_ * H_), 256, 0, stream>>>(qb, kb, attn);
    softmax_rows<<<B_ * H_ * N_, 256, 0, stream>>>(attn);
    attn_av<<<dim3(3, B_ * H_), 256, 0, stream>>>(attn, vbuf, aob);

    gemm_bf16_wmma<2><<<gD, 256, 0, stream>>>(aob, wob, o_b, hidden, nullptr, h1, M_, D_, D_, 1.0f);

    layernorm_to_bf16<<<M_, 256, 0, stream>>>(h1, ln2_g, ln2_b, yb);

    const dim3 gF1((M_ + 127) / 128, FF_ / 64);  // 65 x 64
    gemm_bf16_wmma<1><<<gF1, 256, 0, stream>>>(yb, wf1b, fc1_b, nullptr, ffb, nullptr, M_, D_, FF_, 1.0f);
    gemm_bf16_wmma<2><<<gD,  256, 0, stream>>>(ffb, wf2b, fc2_b, h1, nullptr, outh, M_, FF_, D_, 1.0f);
}
